// GAT_GatedAttn_51917564674535
// MI455X (gfx1250) — compile-verified, matrix operand order checked by reading, not run on hardware
//
#include <hip/hip_runtime.h>
#include <math.h>

#define NN 40000
#define EE 640000
#define DIN 512
#define HC 512
#define HID 128
#define NHEAD 4

typedef __attribute__((ext_vector_type(16))) __bf16 v16bf;
typedef __attribute__((ext_vector_type(8)))  float  v8f;

// ---------------- utility ----------------
__device__ __forceinline__ void atomicMaxF(float* addr, float v) {
    // sign/magnitude trick: works for mixed signs with -inf init
    if (v >= 0.f) atomicMax((int*)addr, __float_as_int(v));
    else          atomicMin((unsigned int*)addr, __float_as_uint(v));
}

__device__ __forceinline__ float lrelu2(float v) { return v > 0.f ? v : 0.2f * v; }

__global__ void fill_kernel(float* __restrict__ p, float v, size_t n) {
    size_t i = (size_t)blockIdx.x * 256 + threadIdx.x;
    if (i < n) p[i] = v;
}

// Transpose + hi/lo bf16 split of a weight: Wthi/Wtlo[n*K+k] = split(W[k*Nout+n])
__global__ void transpose_split(const float* __restrict__ W, __bf16* __restrict__ Wthi,
                                __bf16* __restrict__ Wtlo, int K, int Nout) {
    int idx = blockIdx.x * 256 + threadIdx.x;
    if (idx < K * Nout) {
        int k = idx / Nout, n = idx - k * Nout;
        float x = W[idx];
        __bf16 h = (__bf16)x;
        Wthi[(size_t)n * K + k] = h;
        Wtlo[(size_t)n * K + k] = (__bf16)(x - (float)h);
    }
}

// Streaming hi/lo bf16 split of an activation matrix (row-major preserved)
__global__ void convert_split(const float* __restrict__ src, __bf16* __restrict__ hi,
                              __bf16* __restrict__ lo, size_t n) {
    size_t i = (size_t)blockIdx.x * 256 + threadIdx.x;
    if (i < n) {
        float x = src[i];
        __bf16 h = (__bf16)x;
        hi[i] = h;
        lo[i] = (__bf16)(x - (float)h);
    }
}

// ---------------- WMMA GEMM: Y = act(A[MxK] @ W + bias) --------------------------------
// Operands pre-split into bf16 hi/lo (bf16x3 error compensation ~ fp32 accuracy).
// One wave computes a 16x64 output strip (4 f32 accumulators), K stepped by 32.
// Inner loop: 10 x 32B bf16 loads + 12 v_wmma_f32_16x16x32_bf16, almost no VALU.
__global__ __launch_bounds__(256)
void gemm_bf16x3(const __bf16* __restrict__ Ahi, const __bf16* __restrict__ Alo,
                 const __bf16* __restrict__ Whi, const __bf16* __restrict__ Wlo,
                 const float* __restrict__ bias, float* __restrict__ Y,
                 int M, int K, int Nout, int act) {
    const int wave = threadIdx.x >> 5;
    const int lane = threadIdx.x & 31;
    const int S = Nout >> 6;                 // 64-wide strips
    const int task = blockIdx.x * 8 + wave;
    const int nTasks = (M >> 4) * S;
    if (task >= nTasks) return;              // wave-uniform: EXEC stays all-1s
    const int row_tile = task / S;
    const int strip    = task - row_tile * S;

    const int m     = row_tile * 16 + (lane & 15);
    const int khalf = (lane >> 4) << 4;      // 0 or 16
    const int ncol  = strip * 64 + (lane & 15);

    const __bf16* arow_hi = Ahi + (size_t)m * K + khalf;
    const __bf16* arow_lo = Alo + (size_t)m * K + khalf;

    v8f acc[4] = {};
    for (int k0 = 0; k0 < K; k0 += 32) {
        v16bf ahi = *(const v16bf*)(arow_hi + k0);
        v16bf alo = *(const v16bf*)(arow_lo + k0);
#pragma unroll
        for (int nt = 0; nt < 4; ++nt) {
            size_t boff = (size_t)(ncol + nt * 16) * K + khalf + k0;
            v16bf bhi = *(const v16bf*)(Whi + boff);
            v16bf blo = *(const v16bf*)(Wlo + boff);
            acc[nt] = __builtin_amdgcn_wmma_f32_16x16x32_bf16(
                false, ahi, false, bhi, (short)0, acc[nt], false, false);
            acc[nt] = __builtin_amdgcn_wmma_f32_16x16x32_bf16(
                false, ahi, false, blo, (short)0, acc[nt], false, false);
            acc[nt] = __builtin_amdgcn_wmma_f32_16x16x32_bf16(
                false, alo, false, bhi, (short)0, acc[nt], false, false);
        }
    }
    // D layout: VGPR r, lanes 0-15 -> M=r, lanes 16-31 -> M=r+8; N = lane&15
    const int rbase = row_tile * 16 + ((lane >> 4) << 3);
#pragma unroll
    for (int nt = 0; nt < 4; ++nt) {
        int col = strip * 64 + nt * 16 + (lane & 15);
        float bb = bias ? bias[col] : 0.f;
#pragma unroll
        for (int r = 0; r < 8; ++r) {
            float v = acc[nt][r] + bb;
            if (act == 1)      v = fmaxf(v, 0.f);
            else if (act == 2) v = tanhf(v);
            Y[(size_t)(rbase + r) * Nout + col] = v;
        }
    }
}

// ---------------- per-node attention logits: es/ed = <h[n,h,:], a[h,:]> -------------
__global__ __launch_bounds__(256)
void esed_kernel(const float* __restrict__ HH, const float* __restrict__ as,
                 const float* __restrict__ ad, float* __restrict__ es,
                 float* __restrict__ ed, int n_nodes) {
    int wave = threadIdx.x >> 5, lane = threadIdx.x & 31;
    int n = blockIdx.x * 8 + wave;
    if (n >= n_nodes) return;
    const float* hrow = HH + (size_t)n * HC;
#pragma unroll
    for (int h = 0; h < NHEAD; ++h) {
        float ss = 0.f, dd = 0.f;
        for (int c = lane; c < 128; c += 32) {
            float hv = hrow[h * 128 + c];
            ss += hv * as[h * 128 + c];
            dd += hv * ad[h * 128 + c];
        }
        for (int o = 16; o > 0; o >>= 1) {
            ss += __shfl_down(ss, o);
            dd += __shfl_down(dd, o);
        }
        if (lane == 0) { es[n * 4 + h] = ss; ed[n * 4 + h] = dd; }
    }
}

// ---------------- edge phase (E real edges + N self loops) ----------------
__device__ __forceinline__ void edge_sd(const int* ei, int e, int E_, int n_nodes,
                                        int& s_, int& d_) {
    if (e < E_) { s_ = ei[e]; d_ = ei[E_ + e]; }
    else        { s_ = d_ = e - E_; }
}

__global__ void edge_max(const int* __restrict__ ei, const float* __restrict__ es,
                         const float* __restrict__ ed, float* __restrict__ mseg,
                         int E_, int n_nodes) {
    int e = blockIdx.x * 256 + threadIdx.x;
    if (e >= E_ + n_nodes) return;
    int s_, d_; edge_sd(ei, e, E_, n_nodes, s_, d_);
#pragma unroll
    for (int h = 0; h < NHEAD; ++h) {
        float t = lrelu2(es[s_ * 4 + h] + ed[d_ * 4 + h]);
        atomicMaxF(&mseg[d_ * 4 + h], t);
    }
}

__global__ void edge_expsum(const int* __restrict__ ei, const float* __restrict__ es,
                            const float* __restrict__ ed, const float* __restrict__ mseg,
                            float* __restrict__ sseg, float* __restrict__ exe,
                            int E_, int n_nodes) {
    int e = blockIdx.x * 256 + threadIdx.x;
    if (e >= E_ + n_nodes) return;
    int s_, d_; edge_sd(ei, e, E_, n_nodes, s_, d_);
#pragma unroll
    for (int h = 0; h < NHEAD; ++h) {
        float t = lrelu2(es[s_ * 4 + h] + ed[d_ * 4 + h]);
        float ex = expf(t - mseg[d_ * 4 + h]);
        exe[(size_t)e * 4 + h] = ex;
        atomicAdd(&sseg[d_ * 4 + h], ex);
    }
}

// wave per edge; L2-resident gather + scatter-add of 512 channels
__global__ __launch_bounds__(256)
void edge_agg(const int* __restrict__ ei, const float* __restrict__ exe,
              const float* __restrict__ sseg, const float* __restrict__ HH,
              float* __restrict__ AGG, int E_, int n_nodes) {
    int wave = threadIdx.x >> 5, lane = threadIdx.x & 31;
    int e = blockIdx.x * 8 + wave;
    if (e >= E_ + n_nodes) return;
    int s_, d_; edge_sd(ei, e, E_, n_nodes, s_, d_);
    float w[NHEAD];
#pragma unroll
    for (int h = 0; h < NHEAD; ++h)
        w[h] = exe[(size_t)e * 4 + h] / (sseg[d_ * 4 + h] + 1e-16f);
    const float* hrow = HH + (size_t)s_ * HC;
    float* arow = AGG + (size_t)d_ * HC;
    for (int c = lane; c < HC; c += 32)
        atomicAdd(&arow[c], w[c >> 7] * hrow[c]);
}

// ---------------- batch norm ----------------
__global__ __launch_bounds__(256)
void bn_stats(const float* __restrict__ AGG, const float* __restrict__ b,
              float* __restrict__ stats, int n_nodes) {
    int c0 = threadIdx.x;           // channels c0 and c0+256
    int r0 = blockIdx.x * 256;
    int rend = min(r0 + 256, n_nodes);
    float b0 = b[c0], b1 = b[c0 + 256];
    float s0 = 0, q0 = 0, s1 = 0, q1 = 0;
    for (int r = r0; r < rend; ++r) {
        float v0 = AGG[(size_t)r * HC + c0] + b0;
        float v1 = AGG[(size_t)r * HC + c0 + 256] + b1;
        s0 += v0; q0 += v0 * v0; s1 += v1; q1 += v1 * v1;
    }
    atomicAdd(&stats[c0], s0);        atomicAdd(&stats[512 + c0], q0);
    atomicAdd(&stats[c0 + 256], s1);  atomicAdd(&stats[512 + c0 + 256], q1);
}

__global__ void bn_apply(const float* __restrict__ AGG, const float* __restrict__ b,
                         const float* __restrict__ stats, const float* __restrict__ g,
                         const float* __restrict__ be, float* __restrict__ cur,
                         int n_nodes, int add_res) {
    size_t i = (size_t)blockIdx.x * 256 + threadIdx.x;
    if (i >= (size_t)n_nodes * HC) return;
    int c = (int)(i & (HC - 1));
    float inv_n = 1.f / (float)n_nodes;
    float mu  = stats[c] * inv_n;
    float var = stats[512 + c] * inv_n - mu * mu;
    float v = (AGG[i] + b[c] - mu) * rsqrtf(var + 1e-5f) * g[c] + be[c];
    v = lrelu2(v);
    cur[i] = add_res ? cur[i] + v : v;
}

// ---------------- pooling head ----------------
__global__ __launch_bounds__(256)
void scores_kernel(const float* __restrict__ G1, const float* __restrict__ Wg2,
                   const float* __restrict__ bg2, float* __restrict__ scores,
                   int n_nodes) {
    int wave = threadIdx.x >> 5, lane = threadIdx.x & 31;
    int n = blockIdx.x * 8 + wave;
    if (n >= n_nodes) return;
    float s = 0.f;
    for (int c = lane; c < HID; c += 32) s += G1[(size_t)n * HID + c] * Wg2[c];
    for (int o = 16; o > 0; o >>= 1) s += __shfl_down(s, o);
    if (lane == 0) scores[n] = s + bg2[0];
}

__global__ void reduce_max_kernel(const float* __restrict__ scores,
                                  float* __restrict__ scal, int n) {
    __shared__ float sm[8];
    int i = blockIdx.x * 256 + threadIdx.x;
    int wave = threadIdx.x >> 5, lane = threadIdx.x & 31;
    float v = (i < n) ? scores[i] : -INFINITY;
    for (int o = 16; o > 0; o >>= 1) v = fmaxf(v, __shfl_down(v, o));
    if (lane == 0) sm[wave] = v;
    __syncthreads();
    if (threadIdx.x == 0) {
        float m = sm[0];
        for (int j = 1; j < 8; ++j) m = fmaxf(m, sm[j]);
        atomicMaxF(scal, m);
    }
}

__global__ void exp_z_kernel(const float* __restrict__ scores,
                             const float* __restrict__ scal,
                             float* __restrict__ exn, float* __restrict__ Z, int n) {
    int i = blockIdx.x * 256 + threadIdx.x;
    int lane = threadIdx.x & 31;
    float e = 0.f;
    if (i < n) { e = expf(scores[i] - scal[0]); exn[i] = e; }
    float s = e;
    for (int o = 16; o > 0; o >>= 1) s += __shfl_down(s, o);
    if (lane == 0) atomicAdd(Z, s);
}

__global__ __launch_bounds__(128)
void gated_kernel(const float* __restrict__ F2, const float* __restrict__ exn,
                  float* __restrict__ gated, int n_nodes) {
    int c = threadIdx.x;                    // one of 128 channels
    int r0 = blockIdx.x * 256;
    int rend = min(r0 + 256, n_nodes);
    float acc = 0.f;
    for (int r = r0; r < rend; ++r) acc += exn[r] * F2[(size_t)r * HID + c];
    atomicAdd(&gated[c], acc);
}

__global__ __launch_bounds__(256)
void pooled_kernel(const float* __restrict__ cur, float* __restrict__ pooled,
                   int n_nodes) {
    int c0 = threadIdx.x;                   // channels c0 and c0+256
    int r0 = blockIdx.x * 256;
    int rend = min(r0 + 256, n_nodes);
    float s0 = 0.f, s1 = 0.f;
    for (int r = r0; r < rend; ++r) {
        s0 += cur[(size_t)r * HC + c0];
        s1 += cur[(size_t)r * HC + c0 + 256];
    }
    atomicAdd(&pooled[c0], s0);
    atomicAdd(&pooled[c0 + 256], s1);
}

__global__ __launch_bounds__(128)
void final_kernel(const float* __restrict__ gated, const float* __restrict__ pooled,
                  const float* __restrict__ Z, const float* __restrict__ Wp,
                  const float* __restrict__ bp, float* __restrict__ out, int n_nodes) {
    int c = threadIdx.x;
    float invZ = 1.f / Z[0];
    float invN = 1.f / (float)n_nodes;
    float acc = bp[c];
    for (int j = 0; j < HID; ++j) acc += (gated[j] * invZ) * Wp[j * HID + c];
    for (int j = 0; j < HC; ++j)  acc += (pooled[j] * invN) * Wp[(HID + j) * HID + c];
    out[c] = acc;
}

// ---------------- host orchestration ----------------
extern "C" void kernel_launch(void* const* d_in, const int* in_sizes, int n_in,
                              void* d_out, int out_size, void* d_ws, size_t ws_size,
                              hipStream_t stream) {
    (void)in_sizes; (void)n_in; (void)out_size; (void)ws_size;
    const int N = NN, E = EE, Etot = EE + NN;

    const float* x  = (const float*)d_in[0];
    const int*   ei = (const int*)d_in[1];
    const float* W[3]  = {(const float*)d_in[2],  (const float*)d_in[8],  (const float*)d_in[14]};
    const float* AS[3] = {(const float*)d_in[3],  (const float*)d_in[9],  (const float*)d_in[15]};
    const float* AD[3] = {(const float*)d_in[4],  (const float*)d_in[10], (const float*)d_in[16]};
    const float* Bb[3] = {(const float*)d_in[5],  (const float*)d_in[11], (const float*)d_in[17]};
    const float* Gg[3] = {(const float*)d_in[6],  (const float*)d_in[12], (const float*)d_in[18]};
    const float* Be[3] = {(const float*)d_in[7],  (const float*)d_in[13], (const float*)d_in[19]};
    const float* Wg1 = (const float*)d_in[20]; const float* bg1 = (const float*)d_in[21];
    const float* Wg2 = (const float*)d_in[22]; const float* bg2 = (const float*)d_in[23];
    const float* Wf1 = (const float*)d_in[24]; const float* bf1 = (const float*)d_in[25];
    const float* Wf2 = (const float*)d_in[26]; const float* bf2 = (const float*)d_in[27];
    const float* Wp  = (const float*)d_in[28]; const float* bp  = (const float*)d_in[29];

    char* base = (char*)d_ws;
    size_t off = 0;
    auto alloc = [&](size_t bytes) -> void* {
        void* p = base + off;
        off = (off + bytes + 255) & ~(size_t)255;
        return p;
    };
    float*  cur   = (float*)alloc((size_t)N * HC * 4);    // layer output / residual
    float*  HHb   = (float*)alloc((size_t)N * HC * 4);    // GEMM output h
    float*  AGG   = (float*)alloc((size_t)N * HC * 4);    // aggregation target
    __bf16* Ahi   = (__bf16*)alloc((size_t)N * HC * 2);   // activation hi split
    __bf16* Alo   = (__bf16*)alloc((size_t)N * HC * 2);   // activation lo split
    __bf16* Wthi  = (__bf16*)alloc((size_t)512 * 512 * 2);
    __bf16* Wtlo  = (__bf16*)alloc((size_t)512 * 512 * 2);
    float*  es    = (float*)alloc((size_t)N * 4 * 4);
    float*  ed    = (float*)alloc((size_t)N * 4 * 4);
    float*  mseg  = (float*)alloc((size_t)N * 4 * 4);
    float*  sseg  = (float*)alloc((size_t)N * 4 * 4);
    float*  exe   = (float*)alloc((size_t)Etot * 4 * 4);
    float*  stats = (float*)alloc(1024 * 4);
    float*  exn   = (float*)alloc((size_t)N * 4);
    float*  G1    = (float*)alloc((size_t)N * HID * 4);
    float*  F1    = (float*)alloc((size_t)N * HID * 4);
    float*  F2    = (float*)alloc((size_t)N * HID * 4);
    float*  scal  = (float*)alloc(4);
    float*  Zb    = (float*)alloc(4);
    float*  gated = (float*)alloc(HID * 4);
    float*  pooled= (float*)alloc(HC * 4);

    float* scores = (float*)d_out + HID;   // attn_scores output region

    const dim3 B256(256), B128(128);
    const int gElem    = (int)(((size_t)N * HC + 255) / 256);
    const int gEdgeT   = (Etot + 255) / 256;
    const int gEdgeW   = (Etot + 7) / 8;
    const int gNodeW   = (N + 7) / 8;
    const int gRowChunk= (N + 255) / 256;

    for (int l = 0; l < 3; ++l) {
        const float* Ain = (l == 0) ? x : cur;
        convert_split<<<gElem, B256, 0, stream>>>(Ain, Ahi, Alo, (size_t)N * HC);
        transpose_split<<<(512 * 512 + 255) / 256, B256, 0, stream>>>(W[l], Wthi, Wtlo, 512, 512);
        // tasks = 2500 row-tiles * 8 strips = 20000 waves -> 2500 blocks
        gemm_bf16x3<<<2500, B256, 0, stream>>>(Ahi, Alo, Wthi, Wtlo, nullptr, HHb,
                                               N, 512, 512, 0);
        esed_kernel<<<gNodeW, B256, 0, stream>>>(HHb, AS[l], AD[l], es, ed, N);
        fill_kernel<<<(N * 4 + 255) / 256, B256, 0, stream>>>(mseg, -INFINITY, (size_t)N * 4);
        fill_kernel<<<(N * 4 + 255) / 256, B256, 0, stream>>>(sseg, 0.f, (size_t)N * 4);
        fill_kernel<<<gElem, B256, 0, stream>>>(AGG, 0.f, (size_t)N * HC);
        fill_kernel<<<4, B256, 0, stream>>>(stats, 0.f, 1024);
        edge_max<<<gEdgeT, B256, 0, stream>>>(ei, es, ed, mseg, E, N);
        edge_expsum<<<gEdgeT, B256, 0, stream>>>(ei, es, ed, mseg, sseg, exe, E, N);
        edge_agg<<<gEdgeW, B256, 0, stream>>>(ei, exe, sseg, HHb, AGG, E, N);
        bn_stats<<<gRowChunk, B256, 0, stream>>>(AGG, Bb[l], stats, N);
        bn_apply<<<gElem, B256, 0, stream>>>(AGG, Bb[l], stats, Gg[l], Be[l], cur, N, l > 0 ? 1 : 0);
    }

    // split final h once; reused by both Wg1 and Wf1 GEMMs
    convert_split<<<gElem, B256, 0, stream>>>(cur, Ahi, Alo, (size_t)N * HC);

    // attention scores branch: G1 = tanh(h @ Wg1 + bg1); scores = G1 @ Wg2 + bg2
    transpose_split<<<(512 * HID + 255) / 256, B256, 0, stream>>>(Wg1, Wthi, Wtlo, 512, HID);
    gemm_bf16x3<<<625, B256, 0, stream>>>(Ahi, Alo, Wthi, Wtlo, bg1, G1, N, 512, HID, 2);
    scores_kernel<<<gNodeW, B256, 0, stream>>>(G1, Wg2, bg2, scores, N);

    // softmax over all nodes
    fill_kernel<<<1, B256, 0, stream>>>(scal, -INFINITY, 1);
    fill_kernel<<<1, B256, 0, stream>>>(Zb, 0.f, 1);
    fill_kernel<<<1, B256, 0, stream>>>(gated, 0.f, HID);
    fill_kernel<<<2, B256, 0, stream>>>(pooled, 0.f, HC);
    reduce_max_kernel<<<gRowChunk, B256, 0, stream>>>(scores, scal, N);
    exp_z_kernel<<<gRowChunk, B256, 0, stream>>>(scores, scal, exn, Zb, N);

    // feat branch: F2 = relu(h @ Wf1 + bf1) @ Wf2 + bf2
    transpose_split<<<(512 * HID + 255) / 256, B256, 0, stream>>>(Wf1, Wthi, Wtlo, 512, HID);
    gemm_bf16x3<<<625, B256, 0, stream>>>(Ahi, Alo, Wthi, Wtlo, bf1, F1, N, 512, HID, 1);
    // split F1 (K=128) into the same staging buffers
    convert_split<<<(N * HID + 255) / 256, B256, 0, stream>>>(F1, Ahi, Alo, (size_t)N * HID);
    transpose_split<<<(HID * HID + 255) / 256, B256, 0, stream>>>(Wf2, Wthi, Wtlo, HID, HID);
    gemm_bf16x3<<<625, B256, 0, stream>>>(Ahi, Alo, Wthi, Wtlo, bf2, F2, N, HID, HID, 0);

    // gated sum, mean pool, final projection
    gated_kernel<<<gRowChunk, B128, 0, stream>>>(F2, exn, gated, N);
    pooled_kernel<<<gRowChunk, B256, 0, stream>>>(cur, pooled, N);
    final_kernel<<<1, B128, 0, stream>>>(gated, pooled, Zb, Wp, bp, (float*)d_out, N);
}
